// BertEventSGNN_8693013807315
// MI455X (gfx1250) — matrix operand order repaired and checked by reading, not compiled, not run on hardware
//
#include <hip/hip_runtime.h>

#define B_ 256
#define N_INST_ 64
#define L_ 128
#define H_ 128
#define K_ 40
#define R_ (B_ * K_)          // 10240 rows
#define G_ (3 * H_)           // 384 gate width
#define NW_ (3 * H_ * H_)     // 49152 weight elements per matrix

typedef __attribute__((ext_vector_type(16))) _Float16 v16h;
typedef __attribute__((ext_vector_type(8)))  float    v8f;

// ---------------------------------------------------------------------------
// Weight split: w -> (hi, lo) f16 pair so that hi + lo ~= w exactly to 2^-22.
// ---------------------------------------------------------------------------
__global__ __launch_bounds__(256)
void split_weights_kernel(const float* __restrict__ wih, const float* __restrict__ whh,
                          _Float16* __restrict__ wih_hi, _Float16* __restrict__ wih_lo,
                          _Float16* __restrict__ whh_hi, _Float16* __restrict__ whh_lo)
{
    int idx = blockIdx.x * 256 + threadIdx.x;
    if (idx < NW_) {
        float v = wih[idx];
        _Float16 hi = (_Float16)v;
        wih_hi[idx] = hi;
        wih_lo[idx] = (_Float16)(v - (float)hi);
    } else if (idx < 2 * NW_) {
        int i = idx - NW_;
        float v = whh[i];
        _Float16 hi = (_Float16)v;
        whh_hi[i] = hi;
        whh_lo[i] = (_Float16)(v - (float)hi);
    }
}

// ---------------------------------------------------------------------------
// Stage 1: gather ev_embs (h0) + adjacency einsum (x0), one block per batch.
//   ev[b,i,:] = seq_out[row, t, :],  row = b*64 + nid2rows[b, ev_nids[b,i]]
//   x0[b,j,h] = sum_i adj[b,i,j] * ev[b,i,h] + adj_bias[h]
// ---------------------------------------------------------------------------
__global__ __launch_bounds__(256)
void stage1_kernel(const float* __restrict__ seq_out,
                   const int* __restrict__ target_idxs,
                   const int* __restrict__ nid2rows,
                   const int* __restrict__ ev_nids,
                   const float* __restrict__ adj,
                   const float* __restrict__ adj_bias,
                   float* __restrict__ h0_f32,
                   _Float16* __restrict__ h0_hi, _Float16* __restrict__ h0_lo,
                   _Float16* __restrict__ x0_hi, _Float16* __restrict__ x0_lo)
{
    __shared__ float  evS[K_ * H_];     // 20.5 KB
    __shared__ float  adjS[K_ * K_];    // 6.4 KB
    __shared__ size_t ofsS[K_];

    const int b = blockIdx.x, tid = threadIdx.x;

    if (tid < K_) {
        int j = ev_nids[b * K_ + tid];
        int n = nid2rows[b * N_INST_ + j];           // (B, N_INST, 1) flat
        size_t row = (size_t)b * N_INST_ + (size_t)n;
        int t = target_idxs[row];
        ofsS[tid] = (row * L_ + (size_t)t) * H_;
    }
    for (int idx = tid; idx < K_ * K_; idx += 256)
        adjS[idx] = adj[(size_t)b * K_ * K_ + idx];
    __syncthreads();

    // Gather ev rows; emit h0 in f32 + split-f16 forms.
    for (int idx = tid; idx < K_ * H_; idx += 256) {
        int i = idx >> 7, c = idx & (H_ - 1);
        float v = seq_out[ofsS[i] + (size_t)c];
        evS[idx] = v;
        size_t o = (size_t)(b * K_ + i) * H_ + (size_t)c;
        h0_f32[o] = v;
        _Float16 hi = (_Float16)v;
        h0_hi[o] = hi;
        h0_lo[o] = (_Float16)(v - (float)hi);
    }
    __syncthreads();

    // Einsum: a[j,c] = sum_i adj[i,j] * ev[i,c] + bias[c]; split to f16 pair.
    const int c = tid & (H_ - 1), j0 = tid >> 7;  // 2 columns of j per iter
    for (int j = j0; j < K_; j += 2) {
        float acc = adj_bias[c];
        #pragma unroll 8
        for (int i = 0; i < K_; ++i)
            acc = fmaf(adjS[i * K_ + j], evS[i * H_ + c], acc);
        size_t o = (size_t)(b * K_ + j) * H_ + (size_t)c;
        _Float16 hi = (_Float16)acc;
        x0_hi[o] = hi;
        x0_lo[o] = (_Float16)(acc - (float)hi);
    }
}

// ---------------------------------------------------------------------------
// WMMA fragment loaders (CDNA5 ISA VGPR layouts, wave32).
// ---------------------------------------------------------------------------
// A: 16x32 f16 tile; base = start of 16-row block, row stride H_.
__device__ inline v16h ld_a16x32(const _Float16* __restrict__ base, int k0, int lane)
{
    const int hf = (lane >> 4) & 1, mr = lane & 15;
    const _Float16* row = base + (size_t)mr * H_;
    v16h a;
    #pragma unroll
    for (int v = 0; v < 8; ++v) {
        int kb = k0 + ((v & 4) << 2) + hf * 8 + (v & 3) * 2;   // K pairs per layout
        a[2 * v]     = row[kb];
        a[2 * v + 1] = row[kb + 1];
    }
    return a;
}

// B: 32x16 f16 tile from W (G_ x H_, row-major); B[k][n] = W[n0+n][k0+k].
__device__ inline v16h ld_b32x16(const _Float16* __restrict__ W, int n0, int k0, int lane)
{
    const int hf = (lane >> 4) & 1, n = lane & 15;
    const _Float16* row = W + (size_t)(n0 + n) * H_;
    v16h b;
    #pragma unroll
    for (int v = 0; v < 8; ++v) {
        int kb = k0 + ((v & 4) << 2) + hf * 8 + (v & 3) * 2;
        b[2 * v]     = row[kb];
        b[2 * v + 1] = row[kb + 1];
    }
    return b;
}

// One 16x16 output tile: out = A(16x128) * W^T(128x16) + bias, split-f16 x3 WMMA.
__device__ inline void gemm_tile(const _Float16* __restrict__ Ahi,
                                 const _Float16* __restrict__ Alo,
                                 const _Float16* __restrict__ Whi,
                                 const _Float16* __restrict__ Wlo,
                                 const float* __restrict__ bias,
                                 int n0, int lane, float* __restrict__ outS)
{
    const int hf = (lane >> 4) & 1, n = lane & 15;
    const float bv = bias[n0 + n];
    v8f acc;
    #pragma unroll
    for (int v = 0; v < 8; ++v) acc[v] = bv;

    #pragma unroll
    for (int kc = 0; kc < H_; kc += 32) {
        v16h ah = ld_a16x32(Ahi, kc, lane);
        v16h al = ld_a16x32(Alo, kc, lane);
        v16h bh = ld_b32x16(Whi, n0, kc, lane);
        v16h bl = ld_b32x16(Wlo, n0, kc, lane);
        acc = __builtin_amdgcn_wmma_f32_16x16x32_f16(false, ah, false, bh, (short)0, acc, false, false);
        acc = __builtin_amdgcn_wmma_f32_16x16x32_f16(false, ah, false, bl, (short)0, acc, false, false);
        acc = __builtin_amdgcn_wmma_f32_16x16x32_f16(false, al, false, bh, (short)0, acc, false, false);
    }
    #pragma unroll
    for (int v = 0; v < 8; ++v)
        outS[(v + hf * 8) * G_ + n0 + n] = acc[v];   // C layout: M = v + hf*8, N = n
}

// ---------------------------------------------------------------------------
// Fused GRU pass: one block of 128 threads (4 waves) per 16 rows.
// Wave w computes n-tiles [6w, 6w+6) of both gi and gh into LDS, then fused
// gate math writes new h (f32) + its f16 split for the next pass.
// ---------------------------------------------------------------------------
__global__ __launch_bounds__(128)
void gru_pass_kernel(const _Float16* __restrict__ x_hi, const _Float16* __restrict__ x_lo,
                     const _Float16* __restrict__ h_hi, const _Float16* __restrict__ h_lo,
                     const float* __restrict__ h_in,
                     const _Float16* __restrict__ wih_hi, const _Float16* __restrict__ wih_lo,
                     const _Float16* __restrict__ whh_hi, const _Float16* __restrict__ whh_lo,
                     const float* __restrict__ b_ih, const float* __restrict__ b_hh,
                     float* __restrict__ h_out,
                     _Float16* __restrict__ ho_hi, _Float16* __restrict__ ho_lo)
{
    __shared__ float giS[16 * G_];   // 24 KB
    __shared__ float ghS[16 * G_];   // 24 KB

    const int m0   = blockIdx.x * 16;
    const int lane = threadIdx.x & 31;
    const int wv   = threadIdx.x >> 5;

    const _Float16* Axh = x_hi + (size_t)m0 * H_;
    const _Float16* Axl = x_lo + (size_t)m0 * H_;
    const _Float16* Ahh = h_hi + (size_t)m0 * H_;
    const _Float16* Ahl = h_lo + (size_t)m0 * H_;

    for (int t = 0; t < 6; ++t) {
        const int n0 = (wv * 6 + t) * 16;
        gemm_tile(Axh, Axl, wih_hi, wih_lo, b_ih, n0, lane, giS);
        gemm_tile(Ahh, Ahl, whh_hi, whh_lo, b_hh, n0, lane, ghS);
    }
    __syncthreads();

    const int c = threadIdx.x;               // 0..127 = H column
    for (int m = 0; m < 16; ++m) {
        const float* gi = giS + m * G_;
        const float* gh = ghS + m * G_;
        float ir = gi[c], iz = gi[H_ + c], in_ = gi[2 * H_ + c];
        float hr = gh[c], hz = gh[H_ + c], hn  = gh[2 * H_ + c];
        float r = 1.0f / (1.0f + __expf(-(ir + hr)));
        float z = 1.0f / (1.0f + __expf(-(iz + hz)));
        float nn = tanhf(in_ + r * hn);
        size_t o = (size_t)(m0 + m) * H_ + (size_t)c;
        float h = h_in[o];
        float hnew = (1.0f - z) * nn + z * h;
        h_out[o] = hnew;
        _Float16 hi16 = (_Float16)hnew;
        ho_hi[o] = hi16;
        ho_lo[o] = (_Float16)(hnew - (float)hi16);
    }
}

// ---------------------------------------------------------------------------
extern "C" void kernel_launch(void* const* d_in, const int* in_sizes, int n_in,
                              void* d_out, int out_size, void* d_ws, size_t ws_size,
                              hipStream_t stream)
{
    const float* seq_out     = (const float*)d_in[0];
    const int*   target_idxs = (const int*)d_in[1];
    const int*   nid2rows    = (const int*)d_in[2];
    const int*   ev_nids     = (const int*)d_in[3];
    const float* adj         = (const float*)d_in[4];
    const float* adj_bias    = (const float*)d_in[5];
    const float* w_ih        = (const float*)d_in[6];
    const float* w_hh        = (const float*)d_in[7];
    const float* b_ih        = (const float*)d_in[8];
    const float* b_hh        = (const float*)d_in[9];
    float* out = (float*)d_out;

    // Workspace carve-up (256-byte aligned slices).
    char* ws = (char*)d_ws;
    size_t off = 0;
    auto alloc = [&](size_t bytes) -> char* {
        char* p = ws + off;
        off += (bytes + 255) & ~(size_t)255;
        return p;
    };
    const size_t halfRH = (size_t)R_ * H_ * sizeof(_Float16);   // 2.62 MB
    const size_t f32RH  = (size_t)R_ * H_ * sizeof(float);      // 5.24 MB
    const size_t halfW  = (size_t)NW_ * sizeof(_Float16);       // 96 KB

    _Float16* h0_hi = (_Float16*)alloc(halfRH);
    _Float16* h0_lo = (_Float16*)alloc(halfRH);
    _Float16* x0_hi = (_Float16*)alloc(halfRH);
    _Float16* x0_lo = (_Float16*)alloc(halfRH);
    _Float16* h1_hi = (_Float16*)alloc(halfRH);
    _Float16* h1_lo = (_Float16*)alloc(halfRH);
    float*    h0_f  = (float*)alloc(f32RH);
    float*    h1_f  = (float*)alloc(f32RH);
    _Float16* wih_hi = (_Float16*)alloc(halfW);
    _Float16* wih_lo = (_Float16*)alloc(halfW);
    _Float16* whh_hi = (_Float16*)alloc(halfW);
    _Float16* whh_lo = (_Float16*)alloc(halfW);
    (void)ws_size; (void)in_sizes; (void)n_in; (void)out_size;

    split_weights_kernel<<<(2 * NW_ + 255) / 256, 256, 0, stream>>>(
        w_ih, w_hh, wih_hi, wih_lo, whh_hi, whh_lo);

    stage1_kernel<<<B_, 256, 0, stream>>>(
        seq_out, target_idxs, nid2rows, ev_nids, adj, adj_bias,
        h0_f, h0_hi, h0_lo, x0_hi, x0_lo);

    // Pass 1: x = x0, h = h0  ->  h1 (double-buffered: no block reads h1 yet).
    gru_pass_kernel<<<R_ / 16, 128, 0, stream>>>(
        x0_hi, x0_lo, h0_hi, h0_lo, h0_f,
        wih_hi, wih_lo, whh_hi, whh_lo, b_ih, b_hh,
        h1_f, h1_hi, h1_lo);

    // Pass 2: x = h = h1 -> d_out (f32). Split outputs go to dead h0 buffers.
    gru_pass_kernel<<<R_ / 16, 128, 0, stream>>>(
        h1_hi, h1_lo, h1_hi, h1_lo, h1_f,
        wih_hi, wih_lo, whh_hi, whh_lo, b_ih, b_hh,
        out, h0_hi, h0_lo);
}
